// MoE_88021059764414
// MI455X (gfx1250) — matrix operand order repaired and checked by reading, not compiled, not run on hardware
//
#include <hip/hip_runtime.h>
#include <hip/hip_bf16.h>
#include <math.h>

#define D_MODEL 1024
#define HIDDEN  2048
#define E_ROUT  14
#define E_SH    2
#define NTOK    2048
#define EPS_G   1e-9f
#define RMS_EPSV 1.1920929e-07f

typedef __attribute__((ext_vector_type(16))) __bf16 v16bf;
typedef __attribute__((ext_vector_type(8)))  __bf16 v8bf;
typedef __attribute__((ext_vector_type(8)))  float  v8f;
typedef __attribute__((ext_vector_type(4)))  unsigned int u32x4;
typedef __attribute__((ext_vector_type(8)))  int i32x8;
typedef __attribute__((ext_vector_type(4)))  int i32x4;

// Tensor Data Mover available? (device pass only; host pass lacks the builtin)
#if defined(__HIP_DEVICE_COMPILE__) && __has_builtin(__builtin_amdgcn_tensor_load_to_lds) && __has_builtin(__builtin_amdgcn_s_wait_tensorcnt)
#define USE_TDM 1
#else
#define USE_TDM 0
#endif

// ---------------------------------------------------------------- utilities
__global__ void zero_f32_k(float* __restrict__ p, int n) {
  for (int i = blockIdx.x * blockDim.x + threadIdx.x; i < n;
       i += gridDim.x * blockDim.x) p[i] = 0.f;
}
__global__ void zero_i32_k(int* __restrict__ p, int n) {
  for (int i = blockIdx.x * blockDim.x + threadIdx.x; i < n;
       i += gridDim.x * blockDim.x) p[i] = 0;
}

// fp32 [R][C] -> bf16 transposed [C][R], per expert (blockIdx.z), LDS-tiled so
// both the loads and the transposed stores stay coalesced.
__global__ void cvt_t_k(const float* __restrict__ s, __bf16* __restrict__ d,
                        int R, int C) {
  __shared__ float t[32][33];
  const float* se = s + (size_t)blockIdx.z * R * C;
  __bf16*      de = d + (size_t)blockIdx.z * R * C;
  const int c0 = blockIdx.x * 32, r0 = blockIdx.y * 32;
#pragma unroll
  for (int j = 0; j < 4; ++j) {
    const int r = threadIdx.y + j * 8;
    t[r][threadIdx.x] = se[(size_t)(r0 + r) * C + c0 + threadIdx.x];
  }
  __syncthreads();
#pragma unroll
  for (int j = 0; j < 4; ++j) {
    const int c = threadIdx.y + j * 8;
    de[(size_t)(c0 + c) * R + r0 + threadIdx.x] = (__bf16)t[threadIdx.x][c];
  }
}

// --------------------------------------------------- router: softmax + top2
__global__ void router_k(const float* __restrict__ x, const float* __restrict__ rw,
                         int* __restrict__ tk_idx, float* __restrict__ tk_gate,
                         int* __restrict__ counts) {
  __shared__ float sm[E_ROUT * 128];
  const int t = blockIdx.x, tid = threadIdx.x;
  float acc[E_ROUT];
#pragma unroll
  for (int e = 0; e < E_ROUT; ++e) acc[e] = 0.f;
  for (int d = tid; d < D_MODEL; d += 128) {
    const float xv = x[t * D_MODEL + d];
#pragma unroll
    for (int e = 0; e < E_ROUT; ++e) acc[e] += xv * rw[d * E_ROUT + e];
  }
#pragma unroll
  for (int e = 0; e < E_ROUT; ++e) sm[e * 128 + tid] = acc[e];
  __syncthreads();
  for (int s = 64; s > 0; s >>= 1) {
    if (tid < s)
      for (int e = 0; e < E_ROUT; ++e) sm[e * 128 + tid] += sm[e * 128 + tid + s];
    __syncthreads();
  }
  if (tid == 0) {
    float l[E_ROUT];
    float mx = -1e30f;
    for (int e = 0; e < E_ROUT; ++e) { l[e] = sm[e * 128]; mx = fmaxf(mx, l[e]); }
    float se = 0.f;
    for (int e = 0; e < E_ROUT; ++e) { l[e] = expf(l[e] - mx); se += l[e]; }
    const float inv_se = 1.f / se;
    for (int e = 0; e < E_ROUT; ++e) l[e] *= inv_se;
    int i0 = 0;
    for (int e = 1; e < E_ROUT; ++e) if (l[e] > l[i0]) i0 = e;
    int i1 = (i0 == 0) ? 1 : 0;
    for (int e = 0; e < E_ROUT; ++e) if (e != i0 && l[e] > l[i1]) i1 = e;
    const float v0 = l[i0], v1 = l[i1];
    const float inv = 1.f / (v0 + v1 + EPS_G);
    tk_idx[2 * t] = i0;  tk_idx[2 * t + 1] = i1;
    tk_gate[2 * t] = v0 * inv; tk_gate[2 * t + 1] = v1 * inv;
    atomicAdd(&counts[i0], 1); atomicAdd(&counts[i1], 1);
  }
}

__global__ void scan_k(const int* __restrict__ counts, int* __restrict__ offsets,
                       int* __restrict__ cursors) {
  if (threadIdx.x == 0) {
    int s = 0;
    for (int e = 0; e < E_ROUT; ++e) { offsets[e] = s; s += counts[e]; }
  }
  if (threadIdx.x < 16) cursors[threadIdx.x] = 0;
}

__global__ void scatter_k(const int* __restrict__ tk_idx,
                          const float* __restrict__ tk_gate,
                          const int* __restrict__ offsets, int* __restrict__ cursors,
                          int* __restrict__ tok_list, float* __restrict__ gate_list) {
  const int t = blockIdx.x * blockDim.x + threadIdx.x;
  if (t >= NTOK) return;
#pragma unroll
  for (int k = 0; k < 2; ++k) {
    const int e = tk_idx[2 * t + k];
    const int slot = atomicAdd(&cursors[e], 1);
    tok_list[offsets[e] + slot] = t;
    gate_list[offsets[e] + slot] = tk_gate[2 * t + k];
  }
}

// -------------------------------------------- RMS norm (token part) -> bf16
__global__ void rmsnorm_k(const float* __restrict__ x, __bf16* __restrict__ xhat) {
  __shared__ float sm[256];
  const int t = blockIdx.x, tid = threadIdx.x;
  float ss = 0.f;
  for (int d = tid; d < D_MODEL; d += 256) {
    const float v = x[t * D_MODEL + d];
    ss += v * v;
  }
  sm[tid] = ss;
  __syncthreads();
  for (int s = 128; s > 0; s >>= 1) {
    if (tid < s) sm[tid] += sm[tid + s];
    __syncthreads();
  }
  const float inv = rsqrtf(sm[0] / (float)D_MODEL + RMS_EPSV);
  for (int d = tid; d < D_MODEL; d += 256)
    xhat[t * D_MODEL + d] = (__bf16)(x[t * D_MODEL + d] * inv);
}

// ---------------------------------------------------------- TDM descriptor
#if USE_TDM
typedef __attribute__((address_space(3))) const char lds_cchar;
__device__ __forceinline__ unsigned lds_off(const void* p) {
  return (unsigned)(unsigned long long)(lds_cchar*)p;
}
// 2-D tile load: 32 elems (64B) x 64 rows of bf16, row stride `stride` elems.
// LDS pad: every 16 DWORDs (code 3) insert 4 DWORDs (code 3) -> 40-elem rows,
// matching the [64][40] LDS tiles. Rows >= tensor_h are zero-filled by TDM OOB.
__device__ __forceinline__ void tdm_load_tile(unsigned lds, const __bf16* g,
                                              unsigned tensor_h,
                                              unsigned long long stride) {
  const unsigned long long ga = (unsigned long long)g;
  u32x4 g0;
  g0[0] = 1u;                                   // count=1, user mode
  g0[1] = lds;                                  // lds_addr
  g0[2] = (unsigned)ga;                         // global_addr[31:0]
  g0[3] = (unsigned)((ga >> 32) & 0x1FFFFFFu) | (2u << 30);  // [56:32] | type=2
  i32x8 g1;
  g1[0] = (int)((1u << 16) |                    // data_size = 2 bytes
                (1u << 20) |                    // pad_enable
                (3u << 22) |                    // pad_interval: 16 DWORDs
                (3u << 25));                    // pad_amount:   4 DWORDs
  g1[1] = (int)((32u & 0xFFFFu) << 16);         // tensor_dim0[15:0] = 32
  g1[2] = (int)((tensor_h & 0xFFFFu) << 16);    // dim0[31:16]=0 | dim1[15:0]
  g1[3] = (int)(((tensor_h >> 16) & 0xFFFFu) | (32u << 16));  // dim1 hi | tile_dim0=32
  g1[4] = (int)64u;                             // tile_dim1 = 64, tile_dim2 = 0
  g1[5] = (int)(unsigned)(stride & 0xFFFFFFFFull);            // dim0_stride lo
  g1[6] = (int)(unsigned)((stride >> 32) & 0xFFFFull);        // dim0_stride hi
  g1[7] = 0;
  const i32x4 z4 = {0, 0, 0, 0};
#if __clang_major__ >= 23
  const i32x8 z8 = {0, 0, 0, 0, 0, 0, 0, 0};
  __builtin_amdgcn_tensor_load_to_lds(g0, g1, z4, z4, z8, 0);
#else
  __builtin_amdgcn_tensor_load_to_lds(g0, g1, z4, z4, 0);
#endif
}
#endif  // USE_TDM

// --------------------------------------------------------------- GEMM frags
// WMMA bf16 16x16x32 layout (ISA 7.12.2):
//  A: m = lane&15; lanes 0-15 carry K {0..7,16..23}, lanes 16-31 K {8..15,24..31}
//  B (LDS [n][k]): n = lane&15; same K split as A
//  C/D: elem r -> m = r + (lane>>4)*8, n = lane&15
__device__ __forceinline__ void load_afrag(v16bf& f, const __bf16* row0, int lh) {
  const v8bf lo = *(const v8bf*)(row0 + lh * 8);
  const v8bf hi = *(const v8bf*)(row0 + 16 + lh * 8);
#pragma unroll
  for (int j = 0; j < 8; ++j) { f[j] = lo[j]; f[j + 8] = hi[j]; }
}
__device__ __forceinline__ void load_bfrag(v16bf& f, const __bf16* row0, int lh) {
  const v8bf lo = *(const v8bf*)(row0 + lh * 8);
  const v8bf hi = *(const v8bf*)(row0 + 16 + lh * 8);
#pragma unroll
  for (int j = 0; j < 8; ++j) { f[j] = lo[j]; f[j + 8] = hi[j]; }
}

// ----------------------------- GEMM1: [cnt x D] @ [D x H] -> SiLU -> bf16 h
// B source is TRANSPOSED w1 (w1T[e][n=H][k=D]) so tiles are row-major for TDM.
__global__ void __launch_bounds__(128)
gemm1_k(const __bf16* __restrict__ xhat, const float* __restrict__ rms_w,
        const float* __restrict__ rms_w_s, const __bf16* __restrict__ w1T,
        const __bf16* __restrict__ w1sT, const int* __restrict__ counts,
        const int* __restrict__ offsets, const int* __restrict__ tok_list,
        __bf16* __restrict__ h) {
  const int e = blockIdx.z;
  const bool routed = (e < E_ROUT);
  int cnt, off, hbase;
  const float* rw;
  const __bf16* Bt;
  if (routed) {
    cnt = counts[e]; off = offsets[e]; hbase = off;
    rw = rms_w + e * D_MODEL;
    Bt = w1T + (size_t)e * HIDDEN * D_MODEL;
  } else {
    const int s2 = e - E_ROUT;
    cnt = NTOK; off = 0; hbase = 2 * NTOK + s2 * NTOK;
    rw = rms_w_s + s2 * D_MODEL;
    Bt = w1sT + (size_t)s2 * HIDDEN * D_MODEL;
  }
  const int m0 = blockIdx.x * 64;
  if (m0 >= cnt) return;
  const int n0 = blockIdx.y * 64;

  __shared__ __bf16 As[64][40];                  // [m][k]
#if USE_TDM
  __shared__ __bf16 Bs[2][64][40];               // [n][k], double buffered
#else
  __shared__ __bf16 Bs[1][64][40];
#endif

  const int tid = threadIdx.x;
  const int lane = tid & 31, wv = tid >> 5;
  const int wm = (wv >> 1) * 32, wn = (wv & 1) * 32;
  const int lm = lane & 15, lh = lane >> 4;

  v8f acc[2][2];
#pragma unroll
  for (int a = 0; a < 2; ++a)
#pragma unroll
    for (int b = 0; b < 2; ++b)
#pragma unroll
      for (int r = 0; r < 8; ++r) acc[a][b][r] = 0.f;

  const int ar = tid >> 1, ac0 = (tid & 1) * 16;
  const int agr = m0 + ar;
  int atok = -1;
  if (agr < cnt) atok = routed ? tok_list[off + agr] : agr;

#if USE_TDM
  const unsigned ldsB0 = lds_off(&Bs[0][0][0]);
  const unsigned ldsB1 = lds_off(&Bs[1][0][0]);
  if (wv == 0)
    tdm_load_tile(ldsB0, Bt + (size_t)n0 * D_MODEL, 64u, D_MODEL);
#endif

  for (int k0 = 0, st = 0; k0 < D_MODEL; k0 += 32, ++st) {
    const int cur = st & 1;
    // A tile: gathered tokens, fold per-expert RMS weight, fp32->bf16
    if (atok >= 0) {
      const v8bf* px = (const v8bf*)(xhat + (size_t)atok * D_MODEL + k0 + ac0);
#pragma unroll
      for (int p = 0; p < 2; ++p) {
        const v8bf xv = px[p];
        v8bf ov;
#pragma unroll
        for (int j = 0; j < 8; ++j)
          ov[j] = (__bf16)((float)xv[j] * rw[k0 + ac0 + p * 8 + j]);
        *(v8bf*)&As[ar][ac0 + p * 8] = ov;
      }
    } else {
      v8bf z;
#pragma unroll
      for (int j = 0; j < 8; ++j) z[j] = (__bf16)0.f;
      *(v8bf*)&As[ar][ac0] = z;
      *(v8bf*)&As[ar][ac0 + 8] = z;
    }
#if USE_TDM
    if (wv == 0) {
      if (k0 + 32 < D_MODEL) {  // DMA next B tile while this one is consumed
        tdm_load_tile(cur ? ldsB0 : ldsB1,
                      Bt + (size_t)n0 * D_MODEL + k0 + 32, 64u, D_MODEL);
        __builtin_amdgcn_s_wait_tensorcnt((short)1);
      } else {
        __builtin_amdgcn_s_wait_tensorcnt((short)0);
      }
    }
#else
    {
      const v8bf* pb = (const v8bf*)(Bt + (size_t)(n0 + ar) * D_MODEL + k0 + ac0);
      *(v8bf*)&Bs[0][ar][ac0] = pb[0];
      *(v8bf*)&Bs[0][ar][ac0 + 8] = pb[1];
    }
#endif
    __syncthreads();

    v16bf aF[2], bF[2];
#pragma unroll
    for (int mf = 0; mf < 2; ++mf) load_afrag(aF[mf], &As[wm + mf * 16 + lm][0], lh);
#if USE_TDM
#pragma unroll
    for (int nf = 0; nf < 2; ++nf)
      load_bfrag(bF[nf], &Bs[cur][wn + nf * 16 + lm][0], lh);
#else
#pragma unroll
    for (int nf = 0; nf < 2; ++nf)
      load_bfrag(bF[nf], &Bs[0][wn + nf * 16 + lm][0], lh);
#endif
#pragma unroll
    for (int mf = 0; mf < 2; ++mf)
#pragma unroll
      for (int nf = 0; nf < 2; ++nf)
        acc[mf][nf] = __builtin_amdgcn_wmma_f32_16x16x32_bf16(
            false, aF[mf], false, bF[nf], (short)0, acc[mf][nf], false, false);
    __syncthreads();
  }

  // fused SiLU epilogue -> bf16 h
#pragma unroll
  for (int mf = 0; mf < 2; ++mf)
#pragma unroll
    for (int nf = 0; nf < 2; ++nf)
#pragma unroll
      for (int r = 0; r < 8; ++r) {
        const int m = wm + mf * 16 + r + lh * 8;
        const int gr = m0 + m;
        if (gr < cnt) {
          const int col = n0 + wn + nf * 16 + lm;
          float v = acc[mf][nf][r];
          v = v / (1.f + expf(-v));
          h[(size_t)(hbase + gr) * HIDDEN + col] = (__bf16)v;
        }
      }
}

// -------- GEMM2: [cnt x H] @ [H x D], gate-scale + atomic scatter into out
// Both A (h rows) and B (transposed w2) tiles are TDM-fed, double buffered.
__global__ void __launch_bounds__(128)
gemm2_k(const __bf16* __restrict__ h, const __bf16* __restrict__ w2T,
        const __bf16* __restrict__ w2sT, const int* __restrict__ counts,
        const int* __restrict__ offsets, const int* __restrict__ tok_list,
        const float* __restrict__ gate_list, float* __restrict__ out) {
  const int e = blockIdx.z;
  const bool routed = (e < E_ROUT);
  int cnt, off, hbase;
  const __bf16* Bt;
  if (routed) {
    cnt = counts[e]; off = offsets[e]; hbase = off;
    Bt = w2T + (size_t)e * HIDDEN * D_MODEL;
  } else {
    const int s2 = e - E_ROUT;
    cnt = NTOK; off = 0; hbase = 2 * NTOK + s2 * NTOK;
    Bt = w2sT + (size_t)s2 * HIDDEN * D_MODEL;
  }
  const int m0 = blockIdx.x * 64;
  if (m0 >= cnt) return;
  const int n0 = blockIdx.y * 64;

#if USE_TDM
  __shared__ __bf16 As[2][64][40];
  __shared__ __bf16 Bs[2][64][40];
#else
  __shared__ __bf16 As[1][64][40];
  __shared__ __bf16 Bs[1][64][40];
#endif

  const int tid = threadIdx.x;
  const int lane = tid & 31, wv = tid >> 5;
  const int wm = (wv >> 1) * 32, wn = (wv & 1) * 32;
  const int lm = lane & 15, lh = lane >> 4;

  v8f acc[2][2];
#pragma unroll
  for (int a = 0; a < 2; ++a)
#pragma unroll
    for (int b = 0; b < 2; ++b)
#pragma unroll
      for (int r = 0; r < 8; ++r) acc[a][b][r] = 0.f;

  const __bf16* Arow0 = h + (size_t)(hbase + m0) * HIDDEN;
  const unsigned remA = (unsigned)(cnt - m0);  // TDM zero-fills rows >= cnt

#if USE_TDM
  const unsigned ldsA0 = lds_off(&As[0][0][0]), ldsA1 = lds_off(&As[1][0][0]);
  const unsigned ldsB0 = lds_off(&Bs[0][0][0]), ldsB1 = lds_off(&Bs[1][0][0]);
  if (wv == 0) {
    tdm_load_tile(ldsA0, Arow0, remA, HIDDEN);
    tdm_load_tile(ldsB0, Bt + (size_t)n0 * HIDDEN, 64u, HIDDEN);
  }
#else
  const int ar = tid >> 1, ac0 = (tid & 1) * 16;
  const bool arow_ok = ((unsigned)ar < remA);
#endif

  for (int k0 = 0, st = 0; k0 < HIDDEN; k0 += 32, ++st) {
    const int cur = st & 1;
#if USE_TDM
    if (wv == 0) {
      if (k0 + 32 < HIDDEN) {
        const unsigned la = cur ? ldsA0 : ldsA1;
        const unsigned lb = cur ? ldsB0 : ldsB1;
        tdm_load_tile(la, Arow0 + k0 + 32, remA, HIDDEN);
        tdm_load_tile(lb, Bt + (size_t)n0 * HIDDEN + k0 + 32, 64u, HIDDEN);
        __builtin_amdgcn_s_wait_tensorcnt((short)2);
      } else {
        __builtin_amdgcn_s_wait_tensorcnt((short)0);
      }
    }
#else
    if (arow_ok) {
      const v8bf* px = (const v8bf*)(Arow0 + (size_t)ar * HIDDEN + k0 + ac0);
      *(v8bf*)&As[0][ar][ac0] = px[0];
      *(v8bf*)&As[0][ar][ac0 + 8] = px[1];
    } else {
      v8bf z;
#pragma unroll
      for (int j = 0; j < 8; ++j) z[j] = (__bf16)0.f;
      *(v8bf*)&As[0][ar][ac0] = z;
      *(v8bf*)&As[0][ar][ac0 + 8] = z;
    }
    {
      const v8bf* pb = (const v8bf*)(Bt + (size_t)(n0 + ar) * HIDDEN + k0 + ac0);
      *(v8bf*)&Bs[0][ar][ac0] = pb[0];
      *(v8bf*)&Bs[0][ar][ac0 + 8] = pb[1];
    }
#endif
    __syncthreads();

    v16bf aF[2], bF[2];
#if USE_TDM
#pragma unroll
    for (int mf = 0; mf < 2; ++mf)
      load_afrag(aF[mf], &As[cur][wm + mf * 16 + lm][0], lh);
#pragma unroll
    for (int nf = 0; nf < 2; ++nf)
      load_bfrag(bF[nf], &Bs[cur][wn + nf * 16 + lm][0], lh);
#else
#pragma unroll
    for (int mf = 0; mf < 2; ++mf)
      load_afrag(aF[mf], &As[0][wm + mf * 16 + lm][0], lh);
#pragma unroll
    for (int nf = 0; nf < 2; ++nf)
      load_bfrag(bF[nf], &Bs[0][wn + nf * 16 + lm][0], lh);
#endif
#pragma unroll
    for (int mf = 0; mf < 2; ++mf)
#pragma unroll
      for (int nf = 0; nf < 2; ++nf)
        acc[mf][nf] = __builtin_amdgcn_wmma_f32_16x16x32_bf16(
            false, aF[mf], false, bF[nf], (short)0, acc[mf][nf], false, false);
    __syncthreads();
  }

#pragma unroll
  for (int mf = 0; mf < 2; ++mf)
#pragma unroll
    for (int nf = 0; nf < 2; ++nf)
#pragma unroll
      for (int r = 0; r < 8; ++r) {
        const int m = wm + mf * 16 + r + lh * 8;
        const int gr = m0 + m;
        if (gr < cnt) {
          int tok; float g;
          if (routed) { tok = tok_list[off + gr]; g = gate_list[off + gr]; }
          else        { tok = gr;                 g = 1.f; }
          const int col = n0 + wn + nf * 16 + lm;
          atomicAdd(&out[(size_t)tok * D_MODEL + col], acc[mf][nf][r] * g);
        }
      }
}

// ------------------------------------------------------------------- launch
extern "C" void kernel_launch(void* const* d_in, const int* in_sizes, int n_in,
                              void* d_out, int out_size, void* d_ws, size_t ws_size,
                              hipStream_t stream) {
  (void)in_sizes; (void)n_in; (void)out_size; (void)ws_size;
  const float* x        = (const float*)d_in[0];
  const float* router_w = (const float*)d_in[1];
  const float* rms_w    = (const float*)d_in[2];
  const float* w1       = (const float*)d_in[3];
  const float* w2       = (const float*)d_in[4];
  const float* rms_w_s  = (const float*)d_in[5];
  const float* w1_s     = (const float*)d_in[6];
  const float* w2_s     = (const float*)d_in[7];
  float* out = (float*)d_out;

  // workspace carving (256B aligned). Total ~172 MB.
  char* base = (char*)d_ws;
  size_t cur = 0;
  auto carve = [&](size_t bytes) {
    cur = (cur + 255) & ~(size_t)255;
    void* p = base + cur;
    cur += bytes;
    return p;
  };
  int*    counts    = (int*)carve(16 * sizeof(int));
  int*    cursors   = (int*)carve(16 * sizeof(int));
  int*    offsets   = (int*)carve(16 * sizeof(int));
  int*    tk_idx    = (int*)carve((size_t)2 * NTOK * sizeof(int));
  float*  tk_gate   = (float*)carve((size_t)2 * NTOK * sizeof(float));
  int*    tok_list  = (int*)carve((size_t)2 * NTOK * sizeof(int));
  float*  gate_list = (float*)carve((size_t)2 * NTOK * sizeof(float));
  __bf16* xhat      = (__bf16*)carve((size_t)NTOK * D_MODEL * 2);
  __bf16* w1T       = (__bf16*)carve((size_t)E_ROUT * D_MODEL * HIDDEN * 2);
  __bf16* w2T       = (__bf16*)carve((size_t)E_ROUT * HIDDEN * D_MODEL * 2);
  __bf16* w1sT      = (__bf16*)carve((size_t)E_SH * D_MODEL * HIDDEN * 2);
  __bf16* w2sT      = (__bf16*)carve((size_t)E_SH * HIDDEN * D_MODEL * 2);
  __bf16* hbuf      = (__bf16*)carve((size_t)4 * NTOK * HIDDEN * 2);

  // 0) zero output accumulator and router counts
  zero_f32_k<<<2048, 256, 0, stream>>>(out, NTOK * D_MODEL);
  zero_i32_k<<<1, 64, 0, stream>>>(counts, 16);

  // 1) routing
  router_k<<<NTOK, 128, 0, stream>>>(x, router_w, tk_idx, tk_gate, counts);
  scan_k<<<1, 32, 0, stream>>>(counts, offsets, cursors);
  scatter_k<<<NTOK / 256, 256, 0, stream>>>(tk_idx, tk_gate, offsets, cursors,
                                            tok_list, gate_list);

  // 2) RMS-norm (token part) -> bf16 activations
  rmsnorm_k<<<NTOK, 256, 0, stream>>>(x, xhat);

  // 3) fp32 -> bf16 transposed weight conversion: w1 -> [H][D], w2 -> [D][H]
  //    (makes every GEMM B tile a row-major 64x32 block = native TDM tile)
  cvt_t_k<<<dim3(HIDDEN / 32, D_MODEL / 32, E_ROUT), dim3(32, 8), 0, stream>>>(
      w1, w1T, D_MODEL, HIDDEN);
  cvt_t_k<<<dim3(D_MODEL / 32, HIDDEN / 32, E_ROUT), dim3(32, 8), 0, stream>>>(
      w2, w2T, HIDDEN, D_MODEL);
  cvt_t_k<<<dim3(HIDDEN / 32, D_MODEL / 32, E_SH), dim3(32, 8), 0, stream>>>(
      w1_s, w1sT, D_MODEL, HIDDEN);
  cvt_t_k<<<dim3(D_MODEL / 32, HIDDEN / 32, E_SH), dim3(32, 8), 0, stream>>>(
      w2_s, w2sT, HIDDEN, D_MODEL);

  // 4) expert FFN: GEMM1+SiLU, then GEMM2+gate+scatter (14 routed + 2 shared)
  gemm1_k<<<dim3(NTOK / 64, HIDDEN / 64, E_ROUT + E_SH), 128, 0, stream>>>(
      xhat, rms_w, rms_w_s, w1T, w1sT, counts, offsets, tok_list, hbuf);
  gemm2_k<<<dim3(NTOK / 64, D_MODEL / 64, E_ROUT + E_SH), 128, 0, stream>>>(
      hbuf, w2T, w2sT, counts, offsets, tok_list, gate_list, out);
}
